// WindowMultiHeadAttention_27805618274641
// MI455X (gfx1250) — matrix operand order repaired and checked by reading, not compiled, not run on hardware
//
#include <hip/hip_runtime.h>

typedef _Float16 f16;
typedef _Float16 v16h __attribute__((ext_vector_type(16)));
typedef _Float16 v8h  __attribute__((ext_vector_type(8)));
typedef float    v8f  __attribute__((ext_vector_type(8)));
typedef float    v4f  __attribute__((ext_vector_type(4)));

#define TT   64      // tokens per window (8x8)
#define CCH  256     // channels
#define NH   8       // heads
#define LDX  264     // row stride (halves) for [64 x 256] tiles: 16B-aligned rows, conflict-free
#define LDV  72      // row stride (halves) for K^T [256 x 64]
#define LDP  72      // row stride (halves) for P   [64 x 64]

__device__ __forceinline__ v8f wmma16(v16h a, v16h b, v8f c) {
  // D = A(16x32 f16) * B(32x16 f16) + C(16x16 f32)
  return __builtin_amdgcn_wmma_f32_16x16x32_f16(false, a, false, b, (short)0, c, false, false);
}

// A fragment: 16x32 tile from row-major storage. lane->row M; K split:
// lanes 0-15 hold K {0..7,16..23}, lanes 16-31 hold K {8..15,24..31}.
__device__ __forceinline__ v16h load_fragA(const f16* base, int ld, int lane) {
  int r  = lane & 15;
  int hi = lane >> 4;
  const f16* p = base + r * ld + hi * 8;
  v8h a0 = *(const v8h*)(p);
  v8h a1 = *(const v8h*)(p + 16);
  return __builtin_shufflevector(a0, a1, 0,1,2,3,4,5,6,7,8,9,10,11,12,13,14,15);
}

// B fragment: 32x16 (KxN) tile from row-major (N contiguous) storage.
// lane l holds column N=l%16; lanes 0-15 hold K {0..7,16..23}, lanes 16-31 hold K {8..15,24..31}.
__device__ __forceinline__ v16h load_fragB(const f16* base, int ld, int lane) {
  const f16* p = base + (lane & 15);
  int hi = lane >> 4;
  v16h b;
  #pragma unroll
  for (int i = 0; i < 8; ++i) {
    b[i]     = p[(hi * 8 + i) * ld];
    b[8 + i] = p[(16 + hi * 8 + i) * ld];
  }
  return b;
}

// ---------------------------------------------------------------------------
// Prep kernel 1: f32 -> f16 weight copies (kept in original [in][out] layout,
// which is exactly the row-major KxN B-matrix layout WMMA wants).
__global__ void prep_weights_kernel(const float* __restrict__ wq,
                                    const float* __restrict__ wkv,
                                    const float* __restrict__ wp,
                                    f16* __restrict__ wq16,
                                    f16* __restrict__ wkv16,
                                    f16* __restrict__ wp16) {
  int i = blockIdx.x * blockDim.x + threadIdx.x;
  if (i < CCH * CCH)     wq16[i]  = (f16)wq[i];
  if (i < CCH * 2 * CCH) wkv16[i] = (f16)wkv[i];
  if (i < CCH * CCH)     wp16[i]  = (f16)wp[i];
}

// Prep kernel 2: relative-position bias MLP -> table [head][64][64] f32.
__global__ void prep_bias_kernel(const float* __restrict__ m1w,
                                 const float* __restrict__ m1b,
                                 const float* __restrict__ m2w,
                                 const float* __restrict__ m2b,
                                 float* __restrict__ biasTab) {
  int p = blockIdx.x * blockDim.x + threadIdx.x;
  if (p >= TT * TT) return;
  int p1 = p >> 6, p2 = p & 63;
  float dy = (float)((p1 >> 3) - (p2 >> 3));
  float dx = (float)((p1 & 7)  - (p2 & 7));
  float ry = copysignf(log1pf(fabsf(dy)), dy);
  float rx = copysignf(log1pf(fabsf(dx)), dx);
  float acc[NH];
  #pragma unroll
  for (int hh = 0; hh < NH; ++hh) acc[hh] = m2b[hh];
  for (int j = 0; j < 256; ++j) {
    float hv = fmaxf(ry * m1w[j] + rx * m1w[256 + j] + m1b[j], 0.0f);
    #pragma unroll
    for (int hh = 0; hh < NH; ++hh) acc[hh] += hv * m2w[j * NH + hh];
  }
  #pragma unroll
  for (int hh = 0; hh < NH; ++hh) biasTab[hh * (TT * TT) + p] = acc[hh];
}

// ---------------------------------------------------------------------------
// Main kernel: one workgroup (8 wave32) per batch element.
__launch_bounds__(256)
__global__ void wmha_kernel(const float* __restrict__ x,
                            const float* __restrict__ kin,
                            const float* __restrict__ biasTab,
                            const f16* __restrict__ wq16,
                            const f16* __restrict__ wkv16,
                            const f16* __restrict__ wp16,
                            const float* __restrict__ bq,
                            const float* __restrict__ bkv,
                            const float* __restrict__ bp,
                            const float* __restrict__ tau,
                            float* __restrict__ out) {
  // Q (phases 1-3) then reused for attention output O (phases 4-5)
  __shared__ __align__(16) f16 sQO[TT * LDX];          // 33792 B
  __shared__ __align__(16) f16 sKT[CCH * LDV];         // K^T  [c][t]   36864 B
  __shared__ __align__(16) f16 sV [TT * LDX];          // V    [t][c]   33792 B
  __shared__ __align__(16) float qn[NH * TT];          // 2048 B
  __shared__ __align__(16) float kn[NH * TT];          // 2048 B
  __shared__ __align__(16) union SmemU {
    f16 xin[2][TT * LDX];   // phases 0-2: x / key_in tiles (67584 B)
    f16 p[NH][TT * LDP];    // phases 3-4: softmax probabilities (73728 B)
  } u;

  const int tid   = threadIdx.x;
  const int lane  = tid & 31;
  const int wave  = tid >> 5;
  const int nlane = lane & 15;
  const int hi8   = (lane >> 4) << 3;
  const int b     = blockIdx.x;

  const float* xb = x   + (size_t)b * (CCH * TT);
  const float* kb = kin + (size_t)b * (CCH * TT);

  // ---- Phase 0: global [C][T] f32 -> LDS [T][C] f16 (coalesced over T) ----
  for (int i = tid; i < CCH * TT; i += 256) {
    int c = i >> 6;
    int t = i & 63;
    u.xin[0][t * LDX + c] = (f16)xb[i];
    u.xin[1][t * LDX + c] = (f16)kb[i];
  }
  __syncthreads();

  // ---- Phase 1: Q = X * Wq + bq -> sQO ----
  for (int s = 0; s < 2; ++s) {
    int nt = wave + 8 * s;                           // output-column strip
    v8f zero = {};
    v8f acc[4] = { zero, zero, zero, zero };
    #pragma unroll 2
    for (int kk = 0; kk < CCH; kk += 32) {
      v16h bf = load_fragB(wq16 + kk * CCH + nt * 16, CCH, lane);
      #pragma unroll
      for (int mt = 0; mt < 4; ++mt) {
        v16h af = load_fragA(&u.xin[0][(mt * 16) * LDX + kk], LDX, lane);
        acc[mt] = wmma16(af, bf, acc[mt]);
      }
    }
    int col = nt * 16 + nlane;
    float bqv = bq[col];
    #pragma unroll
    for (int mt = 0; mt < 4; ++mt)
      #pragma unroll
      for (int r = 0; r < 8; ++r)
        sQO[(mt * 16 + hi8 + r) * LDX + col] = (f16)(acc[mt][r] + bqv);
  }

  // ---- Phase 2: [K|V] = Kin * Wkv + bkv ; K stored transposed, V row-major ----
  for (int s = 0; s < 4; ++s) {
    int nt = wave + 8 * s;                           // 0..31 (cols 0..511)
    v8f zero = {};
    v8f acc[4] = { zero, zero, zero, zero };
    #pragma unroll 2
    for (int kk = 0; kk < CCH; kk += 32) {
      v16h bf = load_fragB(wkv16 + kk * (2 * CCH) + nt * 16, 2 * CCH, lane);
      #pragma unroll
      for (int mt = 0; mt < 4; ++mt) {
        v16h af = load_fragA(&u.xin[1][(mt * 16) * LDX + kk], LDX, lane);
        acc[mt] = wmma16(af, bf, acc[mt]);
      }
    }
    int ncol = nt * 16 + nlane;                      // 0..511, wave-uniform branch below
    float bv = bkv[ncol];
    if (ncol < CCH) {                                // K -> sKT[c][t] (vector store)
      #pragma unroll
      for (int mt = 0; mt < 4; ++mt) {
        v8h pk;
        #pragma unroll
        for (int r = 0; r < 8; ++r) pk[r] = (f16)(acc[mt][r] + bv);
        *(v8h*)(sKT + ncol * LDV + mt * 16 + hi8) = pk;
      }
    } else {                                         // V -> sV[t][c]
      int vc = ncol - CCH;
      #pragma unroll
      for (int mt = 0; mt < 4; ++mt)
        #pragma unroll
        for (int r = 0; r < 8; ++r)
          sV[(mt * 16 + hi8 + r) * LDX + vc] = (f16)(acc[mt][r] + bv);
    }
  }
  __syncthreads();

  // ---- Per-(head,row) L2 norms of q and k ----
  for (int i = tid; i < NH * TT; i += 256) {
    int hh = i >> 6;
    int t  = i & 63;
    const f16* pq = sQO + t * LDX + hh * 32;
    float sq = 0.f, sk = 0.f;
    #pragma unroll
    for (int j = 0; j < 32; ++j) { float a = (float)pq[j]; sq += a * a; }
    const f16* pk = sKT + (hh * 32) * LDV + t;
    #pragma unroll
    for (int j = 0; j < 32; ++j) { float a = (float)pk[j * LDV]; sk += a * a; }
    qn[i] = sqrtf(sq);
    kn[i] = sqrtf(sk);
  }
  __syncthreads();

  // ---- Phase 3: one wave per head: S = Q K^T, cosine-scale, +bias, softmax -> u.p ----
  {
    const int h = wave;
    const float tv = fmaxf(tau[h], 0.01f);
    const float* bias_h = biasTab + h * (TT * TT);
    v16h kb4[4];
    #pragma unroll
    for (int nt = 0; nt < 4; ++nt)
      kb4[nt] = load_fragB(sKT + (h * 32) * LDV + nt * 16, LDV, lane);
    float knv[4];
    #pragma unroll
    for (int nt = 0; nt < 4; ++nt) knv[nt] = kn[h * 64 + nt * 16 + nlane];

    for (int mt = 0; mt < 4; ++mt) {
      v16h qf = load_fragA(sQO + (mt * 16) * LDX + h * 32, LDX, lane);
      float sreg[4][8];
      #pragma unroll
      for (int nt = 0; nt < 4; ++nt) {
        v8f z = {};
        v8f sv2 = wmma16(qf, kb4[nt], z);
        #pragma unroll
        for (int r = 0; r < 8; ++r) sreg[nt][r] = sv2[r];
      }
      float qv[8];
      #pragma unroll
      for (int r = 0; r < 8; ++r) qv[r] = qn[h * 64 + mt * 16 + hi8 + r];
      float rmax[8];
      #pragma unroll
      for (int r = 0; r < 8; ++r) rmax[r] = -3.0e38f;
      #pragma unroll
      for (int nt = 0; nt < 4; ++nt)
        #pragma unroll
        for (int r = 0; r < 8; ++r) {
          float denom = fmaxf(qv[r] * knv[nt], 1e-6f);
          float val = sreg[nt][r] / (denom * tv)
                    + bias_h[(mt * 16 + hi8 + r) * TT + nt * 16 + nlane];
          sreg[nt][r] = val;
          rmax[r] = fmaxf(rmax[r], val);
        }
      // row max / sum: rows live in a 16-lane half -> xor masks 1,2,4,8 stay in-half
      #pragma unroll
      for (int m = 1; m <= 8; m <<= 1)
        #pragma unroll
        for (int r = 0; r < 8; ++r)
          rmax[r] = fmaxf(rmax[r], __shfl_xor(rmax[r], m, 32));
      float rsum[8];
      #pragma unroll
      for (int r = 0; r < 8; ++r) rsum[r] = 0.f;
      #pragma unroll
      for (int nt = 0; nt < 4; ++nt)
        #pragma unroll
        for (int r = 0; r < 8; ++r) {
          float e = __expf(sreg[nt][r] - rmax[r]);
          sreg[nt][r] = e;
          rsum[r] += e;
        }
      #pragma unroll
      for (int m = 1; m <= 8; m <<= 1)
        #pragma unroll
        for (int r = 0; r < 8; ++r)
          rsum[r] += __shfl_xor(rsum[r], m, 32);
      f16* ph = &u.p[h][0];
      #pragma unroll
      for (int r = 0; r < 8; ++r) {
        float inv = 1.0f / rsum[r];
        #pragma unroll
        for (int nt = 0; nt < 4; ++nt)
          ph[(mt * 16 + hi8 + r) * LDP + nt * 16 + nlane] = (f16)(sreg[nt][r] * inv);
      }
    }
  }
  __syncthreads();

  // ---- Phase 4: O_h = P_h * V_h -> sQO (Q is dead, reuse as O) ----
  {
    const int h = wave;
    const f16* ph = &u.p[h][0];
    v16h vb[2][2];
    #pragma unroll
    for (int ntv = 0; ntv < 2; ++ntv)
      #pragma unroll
      for (int kq = 0; kq < 2; ++kq)
        vb[ntv][kq] = load_fragB(sV + (kq * 32) * LDX + h * 32 + ntv * 16, LDX, lane);
    for (int mt = 0; mt < 4; ++mt) {
      v16h pa0 = load_fragA(ph + (mt * 16) * LDP + 0,  LDP, lane);
      v16h pa1 = load_fragA(ph + (mt * 16) * LDP + 32, LDP, lane);
      #pragma unroll
      for (int ntv = 0; ntv < 2; ++ntv) {
        v8f acc = {};
        acc = wmma16(pa0, vb[ntv][0], acc);
        acc = wmma16(pa1, vb[ntv][1], acc);
        int col = h * 32 + ntv * 16 + nlane;
        #pragma unroll
        for (int r = 0; r < 8; ++r)
          sQO[(mt * 16 + hi8 + r) * LDX + col] = (f16)acc[r];
      }
    }
  }
  __syncthreads();

  // ---- Phase 5: out = O * Wp + bp, stored transposed to [B][C][T] ----
  for (int s = 0; s < 2; ++s) {
    int nt = wave + 8 * s;
    v8f zero = {};
    v8f acc[4] = { zero, zero, zero, zero };
    #pragma unroll 2
    for (int kk = 0; kk < CCH; kk += 32) {
      v16h bf = load_fragB(wp16 + kk * CCH + nt * 16, CCH, lane);
      #pragma unroll
      for (int mt = 0; mt < 4; ++mt) {
        v16h af = load_fragA(sQO + (mt * 16) * LDX + kk, LDX, lane);
        acc[mt] = wmma16(af, bf, acc[mt]);
      }
    }
    int col = nt * 16 + nlane;                 // output channel
    float bpv = bp[col];
    float* ob = out + (size_t)b * (CCH * TT) + col * TT;
    #pragma unroll
    for (int mt = 0; mt < 4; ++mt) {           // lane's 8 rows are contiguous t's
      v4f v0, v1;
      #pragma unroll
      for (int r = 0; r < 4; ++r) { v0[r] = acc[mt][r] + bpv; v1[r] = acc[mt][r + 4] + bpv; }
      *(v4f*)(ob + mt * 16 + hi8)     = v0;
      *(v4f*)(ob + mt * 16 + hi8 + 4) = v1;
    }
  }
}

// ---------------------------------------------------------------------------
extern "C" void kernel_launch(void* const* d_in, const int* in_sizes, int n_in,
                              void* d_out, int out_size, void* d_ws, size_t ws_size,
                              hipStream_t stream) {
  const float* x   = (const float*)d_in[0];
  const float* kin = (const float*)d_in[1];
  const float* wq  = (const float*)d_in[2];
  const float* bq  = (const float*)d_in[3];
  const float* wkv = (const float*)d_in[4];
  const float* bkv = (const float*)d_in[5];
  const float* wp  = (const float*)d_in[6];
  const float* bp  = (const float*)d_in[7];
  const float* m1w = (const float*)d_in[8];
  const float* m1b = (const float*)d_in[9];
  const float* m2w = (const float*)d_in[10];
  const float* m2b = (const float*)d_in[11];
  const float* tau = (const float*)d_in[12];

  char* ws = (char*)d_ws;
  float* biasTab = (float*)(ws);            // 8*64*64 f32      = 131072 B
  f16*   wq16    = (f16*)(ws + 131072);     // 256*256 f16      = 131072 B
  f16*   wkv16   = (f16*)(ws + 262144);     // 256*512 f16      = 262144 B
  f16*   wp16    = (f16*)(ws + 524288);     // 256*256 f16      = 131072 B

  prep_weights_kernel<<<512, 256, 0, stream>>>(wq, wkv, wp, wq16, wkv16, wp16);
  prep_bias_kernel<<<16, 256, 0, stream>>>(m1w, m1b, m2w, m2b, biasTab);
  wmha_kernel<<<4096, 256, 0, stream>>>(x, kin, biasTab, wq16, wkv16, wp16,
                                        bq, bkv, bp, tau, (float*)d_out);
}
